// GIB_DS_40261023433010
// MI455X (gfx1250) — matrix operand order repaired and checked by reading, not compiled, not run on hardware
//
#include <hip/hip_runtime.h>
#include <hip/hip_bf16.h>

// ---------------- problem constants (from reference) ----------------
#define HDIM   52          // H
#define FEDIM  10          // FE
#define NSTEPS 3
#define BATCH  256
#define D2     104         // 2H
#define D4     208         // 4H (q_star width per side)
#define G4     416         // 8H (LSTM gate width / final concat width)

// fused-K layout for the edge-message GEMM:
//   kk = k*64 + h, k in [0,53) (52 hid channels + 1 bias channel), h in [0,64) (52 used)
#define KP2     3392       // 53*64 logical K
#define KCH     512        // K staged per chunk (16 KB of f16 in LDS)
#define NCHUNKS 7          // ceil(3392/512) -> staged K = 3584 (zero padded)
#define KBLK    112        // 3584/32 K-blocks of one wmma each
#define NPAD    64         // 4 x 16 N tiles (52 valid cols)
#define BPTOT   (4*KBLK*KCH)   // pre-permuted B: 229376 halves (~459 KB, L2 resident)

typedef __attribute__((ext_vector_type(16))) _Float16 v16h;
typedef __attribute__((ext_vector_type(8)))  _Float16 h8;
typedef __attribute__((ext_vector_type(2)))  _Float16 h2;
typedef __attribute__((ext_vector_type(8)))  float    v8f;

// ---------------- small device helpers ----------------
__device__ __forceinline__ float sigm(float x){ return 1.0f/(1.0f + __expf(-x)); }

// deterministic counter-based uniform in (0,1); same sequence every call
__device__ __forceinline__ float rng_u(unsigned key, unsigned idx){
  unsigned x = idx ^ (key * 0x9E3779B9u);
  x ^= x >> 16; x *= 0x7FEB352Du;
  x ^= x >> 15; x *= 0x846CA68Bu;
  x ^= x >> 16;
  return ((x >> 8) + 0.5f) * (1.0f/16777216.0f);
}

// ---------------- utility kernels ----------------
__global__ void k_zero_f(float* p, int n){
  int i = blockIdx.x*blockDim.x + threadIdx.x;
  if (i < n) p[i] = 0.0f;
}
__global__ void k_zero_i(int* p, int n){
  int i = blockIdx.x*blockDim.x + threadIdx.x;
  if (i < n) p[i] = 0;
}

// generic dense:  Y[n,j] = act( X[n,:] @ W[:, j] + b[j] ),  W row-major [In,Out]
__global__ void k_dense(const float* __restrict__ X, const float* __restrict__ W,
                        const float* __restrict__ b, float* __restrict__ Y,
                        int N, int In, int Out, int relu){
  int i = blockIdx.x*blockDim.x + threadIdx.x;
  if (i >= N*Out) return;
  int n = i / Out, j = i % Out;
  float acc = b ? b[j] : 0.0f;
  const float* xr = X + (size_t)n*In;
  for (int t = 0; t < In; ++t) acc += xr[t]*W[(size_t)t*Out + j];
  Y[i] = relu ? fmaxf(acc, 0.0f) : acc;
}

// ---------------- gather: edge hidden + B-matrix prep ----------------
// hidE[e,k] = relu(ea[e,:]@en1_w[:,k]+en1_b[k]) for k<52 ; hidE[e,52]=1 (bias channel)
__global__ void k_edge_hidden(const float* __restrict__ ea, const float* __restrict__ w,
                              const float* __restrict__ b, float* __restrict__ hidE, int E){
  int i = blockIdx.x*blockDim.x + threadIdx.x;
  if (i >= E*53) return;
  int e = i/53, k = i%53;
  float v = 1.0f;
  if (k < HDIM){
    float acc = b[k];
    #pragma unroll
    for (int f = 0; f < FEDIM; ++f) acc += ea[(size_t)e*FEDIM + f]*w[f*HDIM + k];
    v = fmaxf(acc, 0.0f);
  }
  hidE[i] = v;
}

// Pre-permuted, fragment-major B operand:
//   Bp[((ntile*KBLK + kb)*32 + lane)*16 + j]  holds  B(kk, col) with
//   kk = kb*32 + (lane>=16 ? 16 : 0) + j,  col = ntile*16 + (lane&15)
//   B(kk,col) = en2_w[k, h*52+col] (k<52) | en2_b[h*52+col] (k==52), kk=k*64+h, h<52
__global__ void k_build_Bp(const float* __restrict__ en2w, const float* __restrict__ en2b,
                           _Float16* __restrict__ Bp){
  int i = blockIdx.x*blockDim.x + threadIdx.x;
  if (i >= BPTOT) return;
  int ntile = i / (KBLK*KCH);
  int rem   = i % (KBLK*KCH);
  int kb    = rem / 512;
  int q     = rem % 512;
  int lane  = q / 16, j = q % 16;
  int kk  = kb*32 + ((lane >= 16) ? 16 : 0) + j;
  int col = ntile*16 + (lane & 15);
  float v = 0.0f;
  if (kk < KP2 && col < HDIM){
    int k = kk >> 6, h = kk & 63;
    if (h < HDIM)
      v = (k < HDIM) ? en2w[(size_t)k*(HDIM*HDIM) + h*HDIM + col] : en2b[h*HDIM + col];
  }
  Bp[i] = (_Float16)v;
}

// cnt[n] = #edges with dst==n
__global__ void k_count(const int* __restrict__ dst, float* __restrict__ cnt, int E){
  int e = blockIdx.x*blockDim.x + threadIdx.x;
  if (e < E) atomicAdd(&cnt[dst[e]], 1.0f);
}

// ---------------- THE WMMA KERNEL ----------------
// msg[e,o] = z[e,:] @ B[:,o],  z[e, k*64+h] = hidE[e,k]*nodes[src[e],h]  (rank-1, built on the fly)
// 16 edges/block, 4 waves each own one 16-wide N tile. K is staged into LDS in f16,
// fragment-contiguous, in chunks of 512 -> A frag = 2x ds_load_b128, B frag = 2x global b128.
__global__ __launch_bounds__(128)
void k_msg_gemm(const float* __restrict__ nodes, const float* __restrict__ hidE,
                const _Float16* __restrict__ Bp, const int* __restrict__ src,
                const int* __restrict__ dst, float* __restrict__ agg, int E){
  __shared__ __align__(16) _Float16 sZ[16][KCH];   // 16 KB staged z chunk
  __shared__ h2    sX2[16][32];                    // node features of src, f16, padded to 64
  __shared__ _Float16 sHidH[16][64];               // hid channels, f16, padded to 64
  __shared__ int   sS[16];

  const int eBase = blockIdx.x * 16;
  const int tid   = threadIdx.x;
  if (tid < 16){
    int e = eBase + tid;
    sS[tid] = (e < E) ? src[e] : -1;
  }
  __syncthreads();
  // stage x (as half2 pairs) and hid (as half) once
  for (int idx = tid; idx < 16*32; idx += 128){
    int m = idx >> 5, j = idx & 31;
    float a0 = 0.0f, a1 = 0.0f;
    int s = sS[m];
    if (s >= 0){
      int h0 = 2*j;
      if (h0     < HDIM) a0 = nodes[(size_t)s*HDIM + h0];
      if (h0 + 1 < HDIM) a1 = nodes[(size_t)s*HDIM + h0 + 1];
    }
    h2 p; p[0] = (_Float16)a0; p[1] = (_Float16)a1;
    sX2[m][j] = p;
  }
  for (int idx = tid; idx < 16*64; idx += 128){
    int m = idx >> 6, k = idx & 63;
    float v = 0.0f;
    if ((eBase + m) < E && k < 53) v = hidE[(size_t)(eBase + m)*53 + k];
    sHidH[m][k] = (_Float16)v;
  }

  const int wave = tid >> 5, lane = tid & 31;
  const int lm   = lane & 15;
  const int hi8  = (lane >= 16) ? 8 : 0;    // half the A-frag K offset AND C/D row offset (upper lanes)
  h2 (*sZ2)[KCH/2] = (h2 (*)[KCH/2])sZ;

  v8f acc = {};
  const _Float16* BpT = Bp + (size_t)wave*(KBLK*KCH);   // this wave's N tile

  for (int c = 0; c < NCHUNKS; ++c){
    __syncthreads();
    // cooperative staging of z chunk: 16 rows x 256 half2
    const int kkBase = c*KCH;
    for (int idx = tid; idx < 16*256; idx += 128){
      int m = idx >> 8, p = idx & 255;
      int kk = kkBase + 2*p;
      int kc = kk >> 6;                 // channel (reads 0 beyond 52 via padding)
      int hj = (kk & 63) >> 1;          // half2 index within row
      _Float16 hd = sHidH[m][kc];
      h2 hv; hv[0] = hd; hv[1] = hd;
      sZ2[m][p] = hv * sX2[m][hj];      // v_pk_mul_f16
    }
    if (c + 1 < NCHUNKS)
      __builtin_prefetch(BpT + (size_t)(c + 1)*16*512 + (size_t)lane*16, 0, 1); // global_prefetch_b8
    __syncthreads();

    const _Float16* bptr = BpT + (size_t)c*16*512 + (size_t)lane*16;
    #pragma unroll
    for (int kb = 0; kb < 16; ++kb){
      // A fragment: 16 contiguous halves per lane, k = 16*(lane>=16) + s (matches Bp)
      const h8 lo = *(const h8*)&sZ[lm][kb*32 + 2*hi8];
      const h8 hi = *(const h8*)&sZ[lm][kb*32 + 2*hi8 + 8];
      v16h a;
      #pragma unroll
      for (int i = 0; i < 8; ++i){ a[i] = lo[i]; a[8 + i] = hi[i]; }
      // B fragment: 32 contiguous bytes per lane (fragment-major pre-permuted)
      const h8 blo = *(const h8*)(bptr + (size_t)kb*512);
      const h8 bhi = *(const h8*)(bptr + (size_t)kb*512 + 8);
      v16h b;
      #pragma unroll
      for (int i = 0; i < 8; ++i){ b[i] = blo[i]; b[8 + i] = bhi[i]; }
      acc = __builtin_amdgcn_wmma_f32_16x16x32_f16(false, a, false, b, (short)0, acc, false, false);
    }
  }

  // scatter-add into agg[dst[e], col] (f32 global atomics)
  int col = wave*16 + lm;
  if (col < HDIM){
    #pragma unroll
    for (int r = 0; r < 8; ++r){
      int m = r + hi8;                  // C/D layout: lanes>=16 hold rows M=8..15
      int e = eBase + m;
      if (e < E) atomicAdd(&agg[(size_t)dst[e]*HDIM + col], acc[r]);
    }
  }
}

// newOut[n,j] = [relu(agg/cnt + conv_b) , out] @ msg_w + msg_b
__global__ void k_combine(const float* __restrict__ agg, const float* __restrict__ cnt,
                          const float* __restrict__ convb, const float* __restrict__ outc,
                          const float* __restrict__ mw, const float* __restrict__ mb,
                          float* __restrict__ outn, int N){
  int i = blockIdx.x*blockDim.x + threadIdx.x;
  if (i >= N*HDIM) return;
  int n = i/HDIM, j = i%HDIM;
  float cc = fmaxf(cnt[n], 1.0f);
  float acc = mb[j];
  const float* ar = agg  + (size_t)n*HDIM;
  const float* orow = outc + (size_t)n*HDIM;
  for (int t = 0; t < HDIM; ++t){
    float m = fmaxf(ar[t]/cc + convb[t], 0.0f);
    acc += m * mw[t*HDIM + j];
  }
  for (int t = 0; t < HDIM; ++t) acc += orow[t]*mw[(HDIM + t)*HDIM + j];
  outn[i] = acc;
}

// fu[n,:] = normalize(out[n,:] + x[n,:])
__global__ void k_resnorm(const float* __restrict__ outf, const float* __restrict__ x,
                          float* __restrict__ fu, int N){
  int n = blockIdx.x*blockDim.x + threadIdx.x;
  if (n >= N) return;
  float tmp[HDIM]; float ss = 0.0f;
  for (int j = 0; j < HDIM; ++j){
    float v = outf[(size_t)n*HDIM + j] + x[(size_t)n*HDIM + j];
    tmp[j] = v; ss += v*v;
  }
  float inv = 1.0f / fmaxf(sqrtf(ss), 1e-12f);
  for (int j = 0; j < HDIM; ++j) fu[(size_t)n*HDIM + j] = tmp[j]*inv;
}

// ---------------- batch segment bookkeeping (b indices are sorted) ----------------
__global__ void k_hist(const int* __restrict__ bidx, int* __restrict__ cntB, int N){
  int n = blockIdx.x*blockDim.x + threadIdx.x;
  if (n < N) atomicAdd(&cntB[bidx[n]], 1);
}
__global__ void k_scan(const int* __restrict__ cntB, int* __restrict__ startB){
  if (threadIdx.x == 0){
    int s = 0;
    for (int b = 0; b < BATCH; ++b){ startB[b] = s; s += cntB[b]; }
  }
}

// oprime[i,:] = sum_{j in segB(bidxA[i])} (fa[i,:].fb[j,:]) * fb[j,:]
// (block-diagonal masked GEMM exploited: avoids 64MB `im` + dense 4096^2 work)
__global__ __launch_bounds__(64)
void k_cross(const float* __restrict__ fa, const float* __restrict__ fb,
             const int* __restrict__ bidxA, const int* __restrict__ startB,
             const int* __restrict__ cntB, float* __restrict__ oprime, int Na){
  int i = blockIdx.x; if (i >= Na) return;
  int t = threadIdx.x;
  __shared__ float sa[HDIM];
  __shared__ float accs[HDIM];
  if (t < HDIM){ sa[t] = fa[(size_t)i*HDIM + t]; accs[t] = 0.0f; }
  __syncthreads();
  int b = bidxA[i]; int s = startB[b], c = cntB[b];
  for (int k = t; k < c; k += 64){
    const float* fr = fb + (size_t)(s + k)*HDIM;
    float d = 0.0f;
    for (int j = 0; j < HDIM; ++j) d += sa[j]*fr[j];
    for (int j = 0; j < HDIM; ++j) atomicAdd(&accs[j], d*fr[j]);   // ds_add_f32
  }
  __syncthreads();
  if (t < HDIM) oprime[(size_t)i*HDIM + t] = accs[t];
}

__global__ void k_concat(const float* __restrict__ A, const float* __restrict__ Bc,
                         float* __restrict__ Y, int N, int da, int db){
  int d = da + db;
  int i = blockIdx.x*blockDim.x + threadIdx.x;
  if (i >= N*d) return;
  int n = i/d, j = i%d;
  Y[i] = (j < da) ? A[(size_t)n*da + j] : Bc[(size_t)n*db + (j - da)];
}

// ---------------- compress / bottleneck ----------------
__global__ void k_bnstats(const float* __restrict__ h1, float* __restrict__ mu,
                          float* __restrict__ sq, int N){
  int i = blockIdx.x*blockDim.x + threadIdx.x;
  if (i >= N*HDIM) return;
  float v = h1[i];
  atomicAdd(&mu[i % HDIM], v);
  atomicAdd(&sq[i % HDIM], v*v);
}
__global__ void k_bnfin(float* __restrict__ mu, float* __restrict__ sq, int N){
  int j = threadIdx.x;
  if (j < HDIM){
    float m = mu[j]/(float)N;
    float v = sq[j]/(float)N - m*m;
    mu[j] = m; sq[j] = fmaxf(v, 0.0f);
  }
}
// pl[n] = relu(bn(h1)) @ c2_w + c2_b ; lam = concrete gate ; preserve counter
__global__ void k_plgate(const float* __restrict__ h1, const float* __restrict__ mu,
                         const float* __restrict__ var, const float* __restrict__ g,
                         const float* __restrict__ bb, const float* __restrict__ c2w,
                         const float* __restrict__ c2b, float* __restrict__ pl,
                         float* __restrict__ lam, float* __restrict__ scal,
                         int N, unsigned seed){
  int n = blockIdx.x*blockDim.x + threadIdx.x;
  if (n >= N) return;
  float acc = c2b[0];
  for (int j = 0; j < HDIM; ++j){
    float hb = (h1[(size_t)n*HDIM + j] - mu[j])*rsqrtf(var[j] + 1e-5f)*g[j] + bb[j];
    acc += fmaxf(hb, 0.0f)*c2w[j];
  }
  pl[n] = acc;
  float U   = rng_u(seed, (unsigned)n);
  float eps = -0.9998f*U + 0.9999f;                    // (bias-(1-bias))*U + (1-bias)
  lam[n] = sigm(__logf(eps) - __logf(1.0f - eps) + acc);
  if (sigm(acc) > 0.5f) atomicAdd(&scal[2], 1.0f);
}

// per-batch mean/std of feat (unbiased var as in reference)
__global__ __launch_bounds__(128)
void k_segstat(const float* __restrict__ f2, const int* __restrict__ startB,
               const int* __restrict__ cntB, float* __restrict__ segMean,
               float* __restrict__ segStd){
  int b = blockIdx.x, t = threadIdx.x;
  if (t >= D2) return;
  int s = startB[b], c = cntB[b];
  float sum = 0.0f, ss = 0.0f;
  for (int k = 0; k < c; ++k){
    float v = f2[(size_t)(s + k)*D2 + t];
    sum += v; ss += v*v;
  }
  float cc = fmaxf((float)c, 1.0f);
  float m  = sum/cc;
  float var = (ss - cc*m*m)/fmaxf(cc - 1.0f, 1.0f);
  segMean[b*D2 + t] = m;
  segStd [b*D2 + t] = sqrtf(fmaxf(var, 0.0f));
}

// noisy = nmean + U*nstd; accumulate KL pieces:
// scal[0] = SUM_n mean_d(nstd^2/(std+e)^2), scal[1] = SUM_{n,d} ((nmean-mean)/(std+e))^2
__global__ void k_noisy(const float* __restrict__ f2, const float* __restrict__ lam,
                        const int* __restrict__ bidx, const float* __restrict__ segMean,
                        const float* __restrict__ segStd, float* __restrict__ noisy,
                        float* __restrict__ scal, int N, unsigned seed){
  int i = blockIdx.x*blockDim.x + threadIdx.x;
  if (i >= N*D2) return;
  int n = i/D2, d = i%D2;
  int b = bidx[n];
  float L = lam[n], Ln = 1.0f - L;
  float mean = segMean[b*D2 + d], std = segStd[b*D2 + d];
  float nmean = L*f2[i] + Ln*mean;
  float nstd  = Ln*std;
  noisy[i] = nmean + rng_u(seed, (unsigned)i)*nstd;
  float se = std + 1e-7f;
  float t2 = (nmean - mean)/se;
  atomicAdd(&scal[0], (nstd*nstd)/(se*se)*(1.0f/D2));
  atomicAdd(&scal[1], t2*t2);
}

// ---------------- set2set ----------------
__global__ void k_s2s_gates(const float* __restrict__ qstar, const float* __restrict__ h,
                            const float* __restrict__ wih, const float* __restrict__ bih,
                            const float* __restrict__ whh, const float* __restrict__ bhh,
                            float* __restrict__ gbuf){
  int i = blockIdx.x*blockDim.x + threadIdx.x;
  if (i >= BATCH*G4) return;
  int b = i/G4, u = i%G4;
  float acc = bih[u] + bhh[u];
  const float* q  = qstar + (size_t)b*D4;
  const float* wr = wih   + (size_t)u*D4;
  for (int t = 0; t < D4; ++t) acc += q[t]*wr[t];
  const float* hr  = h   + (size_t)b*D2;
  const float* wr2 = whh + (size_t)u*D2;
  for (int t = 0; t < D2; ++t) acc += hr[t]*wr2[t];
  gbuf[i] = acc;
}
__global__ void k_s2s_update(const float* __restrict__ gbuf, float* __restrict__ h,
                             float* __restrict__ c){
  int i = blockIdx.x*blockDim.x + threadIdx.x;
  if (i >= BATCH*D2) return;
  int b = i/D2, d = i%D2;
  float gi = gbuf[b*G4 + d], gf = gbuf[b*G4 + D2 + d];
  float gg = gbuf[b*G4 + 2*D2 + d], go = gbuf[b*G4 + 3*D2 + d];
  float cn = sigm(gf)*c[i] + sigm(gi)*tanhf(gg);
  c[i] = cn;
  h[i] = sigm(go)*tanhf(cn);
}
__global__ void k_s2s_e(const float* __restrict__ xs, const float* __restrict__ h,
                        const int* __restrict__ bidx, float* __restrict__ evec, int N){
  int n = blockIdx.x*blockDim.x + threadIdx.x;
  if (n >= N) return;
  const float* xr = xs + (size_t)n*D2;
  const float* hr = h  + (size_t)bidx[n]*D2;
  float acc = 0.0f;
  for (int d = 0; d < D2; ++d) acc += xr[d]*hr[d];
  evec[n] = acc;
}
__global__ __launch_bounds__(128)
void k_s2s_attn(const float* __restrict__ xs, const float* __restrict__ evec,
                const int* __restrict__ startB, const int* __restrict__ cntB,
                float* __restrict__ r){
  int b = blockIdx.x, t = threadIdx.x;
  __shared__ float red[128];
  int s = startB[b], c = cntB[b];
  float mx = -3.4e38f;
  for (int k = t; k < c; k += 128) mx = fmaxf(mx, evec[s + k]);
  red[t] = mx; __syncthreads();
  for (int o = 64; o > 0; o >>= 1){ if (t < o) red[t] = fmaxf(red[t], red[t + o]); __syncthreads(); }
  mx = red[0]; __syncthreads();
  float sm = 0.0f;
  for (int k = t; k < c; k += 128) sm += __expf(evec[s + k] - mx);
  red[t] = sm; __syncthreads();
  for (int o = 64; o > 0; o >>= 1){ if (t < o) red[t] += red[t + o]; __syncthreads(); }
  sm = red[0];
  if (t < D2){
    float acc = 0.0f;
    for (int k = 0; k < c; ++k){
      float a = __expf(evec[s + k] - mx)/sm;
      acc += a*xs[(size_t)(s + k)*D2 + t];
    }
    r[b*D2 + t] = (c > 0) ? acc : 0.0f;
  }
}
__global__ void k_s2s_q(const float* __restrict__ h, const float* __restrict__ r,
                        float* __restrict__ qstar){
  int i = blockIdx.x*blockDim.x + threadIdx.x;
  if (i >= BATCH*D4) return;
  int b = i/D4, t = i%D4;
  qstar[i] = (t < D2) ? h[b*D2 + t] : r[b*D2 + (t - D2)];
}

// ---------------- final MLP head + output assembly ----------------
__global__ void k_mlp1(const float* __restrict__ qU, const float* __restrict__ qV,
                       const float* __restrict__ w, const float* __restrict__ b,
                       float* __restrict__ y){
  int i = blockIdx.x*blockDim.x + threadIdx.x;
  if (i >= BATCH*256) return;
  int n = i/256, j = i%256;
  float acc = b[j];
  for (int t = 0; t < D4; ++t) acc += qU[(size_t)n*D4 + t]*w[(size_t)t*256 + j];
  for (int t = 0; t < D4; ++t) acc += qV[(size_t)n*D4 + t]*w[(size_t)(D4 + t)*256 + j];
  y[i] = fmaxf(acc, 0.0f);
}
__global__ void k_mlp3_out(const float* __restrict__ h2v, const float* __restrict__ w,
                           const float* __restrict__ b, const float* __restrict__ scalU,
                           const float* __restrict__ scalV, int Nu, int Nv,
                           float* __restrict__ out, int out_size){
  int n = blockIdx.x*blockDim.x + threadIdx.x;
  if (n < BATCH && n < out_size){
    float acc = b[0];
    for (int t = 0; t < 128; ++t) acc += h2v[(size_t)n*128 + t]*w[t];
    out[n] = acc;
  }
  if (n == 0 && out_size >= BATCH + 4){
    const float inv = 1.0f/(float)(BATCH*D2);
    out[BATCH + 0] = (52.0f*scalU[0] + scalU[1])*inv;   // klu
    out[BATCH + 1] = (52.0f*scalV[0] + scalV[1])*inv;   // klv
    out[BATCH + 2] = scalU[2]/(float)Nu;                // pru
    out[BATCH + 3] = scalV[2]/(float)Nv;                // prv
  }
}

// ---------------- host orchestration ----------------
extern "C" void kernel_launch(void* const* d_in, const int* in_sizes, int n_in,
                              void* d_out, int out_size, void* d_ws, size_t ws_size,
                              hipStream_t stream){
  (void)n_in; (void)ws_size;
  const float* xu  = (const float*)d_in[0];
  const float* eau = (const float*)d_in[1];
  const float* xv  = (const float*)d_in[2];
  const float* eav = (const float*)d_in[3];
  const float* P[44];
  for (int i = 0; i < 44; ++i) P[i] = (const float*)d_in[4 + i];
  const int* eiu = (const int*)d_in[48];
  const int* bu  = (const int*)d_in[49];
  const int* eiv = (const int*)d_in[50];
  const int* bv  = (const int*)d_in[51];
  const int Nu = in_sizes[0]/HDIM, Eu = in_sizes[1]/FEDIM;
  const int Nv = in_sizes[2]/HDIM, Ev = in_sizes[3]/FEDIM;

  // param layout (insertion order in reference dict): 19 per prefix, then head
  // 0 lin0_w 1 lin0_b 2 en1_w 3 en1_b 4 en2_w 5 en2_b 6 conv_b 7 msg_w 8 msg_b
  // 9 s2s_wih 10 s2s_whh 11 s2s_bih 12 s2s_bhh 13 c1_w 14 c1_b 15 bn_g 16 bn_b 17 c2_w 18 c2_b
  const int PU = 0, PV = 19;

  // workspace bump allocator
  char* cur = (char*)d_ws;
  auto CF = [&](size_t n)->float*   { float* p = (float*)cur;     cur += ((n*sizeof(float)+255)&~(size_t)255); return p; };
  auto CI = [&](size_t n)->int*     { int* p = (int*)cur;         cur += ((n*sizeof(int)  +255)&~(size_t)255); return p; };
  auto CH = [&](size_t n)->_Float16*{ _Float16* p=(_Float16*)cur; cur += ((n*2            +255)&~(size_t)255); return p; };

  struct SideWS {
    float *hidE,*out0,*out1,*agg,*cnt,*fu,*oprime,*f2,*h1,*pl,*lam;
    float *segMean,*segStd,*noisy,*evec,*h_s,*c_s,*qstar,*r_s,*mu,*var,*scal;
    _Float16* Bp; int *startB,*cntB;
  };
  auto carve_side = [&](int N, int E)->SideWS{
    SideWS s;
    s.hidE = CF((size_t)E*53);  s.Bp = CH((size_t)BPTOT);
    s.out0 = CF((size_t)N*HDIM); s.out1 = CF((size_t)N*HDIM);
    s.agg  = CF((size_t)N*HDIM); s.cnt  = CF(N);
    s.fu   = CF((size_t)N*HDIM); s.oprime = CF((size_t)N*HDIM);
    s.f2   = CF((size_t)N*D2);   s.h1 = CF((size_t)N*HDIM);
    s.pl = CF(N); s.lam = CF(N);
    s.segMean = CF(BATCH*D2); s.segStd = CF(BATCH*D2);
    s.noisy = CF((size_t)N*D2); s.evec = CF(N);
    s.h_s = CF(BATCH*D2); s.c_s = CF(BATCH*D2);
    s.qstar = CF(BATCH*D4); s.r_s = CF(BATCH*D2);
    s.mu = CF(HDIM); s.var = CF(HDIM); s.scal = CF(4);
    s.startB = CI(BATCH); s.cntB = CI(BATCH);
    return s;
  };
  SideWS su = carve_side(Nu, Eu);
  SideWS sv = carve_side(Nv, Ev);
  float* gbuf = CF(BATCH*G4);
  float* h1m  = CF(BATCH*256);
  float* h2m  = CF(BATCH*128);

  auto gb = [](int n){ return (n + 255)/256; };

  // ---- gather (GNN) for one side ----
  auto run_gather = [&](const float* x, const float* ea, const int* ei, int N, int E,
                        int pp, SideWS& s)->float* {
    const int* src = ei; const int* dst = ei + E;
    k_edge_hidden<<<gb(E*53),256,0,stream>>>(ea, P[pp+2], P[pp+3], s.hidE, E);
    k_build_Bp<<<gb(BPTOT),256,0,stream>>>(P[pp+4], P[pp+5], s.Bp);
    k_dense<<<gb(N*HDIM),256,0,stream>>>(x, P[pp+0], P[pp+1], s.out0, N, HDIM, HDIM, 1);
    k_zero_f<<<gb(N),256,0,stream>>>(s.cnt, N);
    k_count<<<gb(E),256,0,stream>>>(dst, s.cnt, E);
    float* c0 = s.out0; float* c1 = s.out1;
    for (int st = 0; st < NSTEPS; ++st){
      k_zero_f<<<gb(N*HDIM),256,0,stream>>>(s.agg, N*HDIM);
      k_msg_gemm<<<(E + 15)/16,128,0,stream>>>(c0, s.hidE, s.Bp, src, dst, s.agg, E);
      k_combine<<<gb(N*HDIM),256,0,stream>>>(s.agg, s.cnt, P[pp+6], c0, P[pp+7], P[pp+8], c1, N);
      float* t = c0; c0 = c1; c1 = t;
    }
    k_resnorm<<<gb(N),256,0,stream>>>(c0, x, s.fu, N);
    return s.fu;
  };

  float* fu = run_gather(xu, eau, eiu, Nu, Eu, PU, su);
  float* fv = run_gather(xv, eav, eiv, Nv, Ev, PV, sv);

  // ---- batch segment offsets ----
  auto run_seg = [&](const int* bidx, int N, SideWS& s){
    k_zero_i<<<1,256,0,stream>>>(s.cntB, BATCH);
    k_hist<<<gb(N),256,0,stream>>>(bidx, s.cntB, N);
    k_scan<<<1,32,0,stream>>>(s.cntB, s.startB);
  };
  run_seg(bu, Nu, su);
  run_seg(bv, Nv, sv);

  // ---- block-diagonal cross term ----
  k_cross<<<Nu,64,0,stream>>>(fu, fv, bu, sv.startB, sv.cntB, su.oprime, Nu); // u' = im@fv
  k_cross<<<Nv,64,0,stream>>>(fv, fu, bv, su.startB, su.cntB, sv.oprime, Nv); // v' = im.T@fu
  k_concat<<<gb(Nu*D2),256,0,stream>>>(fu, su.oprime, su.f2, Nu, HDIM, HDIM);
  k_concat<<<gb(Nv*D2),256,0,stream>>>(fv, sv.oprime, sv.f2, Nv, HDIM, HDIM);

  // ---- bottleneck + set2set for one side ----
  auto run_bottleneck = [&](const int* bidx, int N, int pp, SideWS& s,
                            unsigned seedGate, unsigned seedNoise){
    k_zero_f<<<1,32,0,stream>>>(s.scal, 4);
    // compress
    k_dense<<<gb(N*HDIM),256,0,stream>>>(s.f2, P[pp+13], P[pp+14], s.h1, N, D2, HDIM, 0);
    k_zero_f<<<1,64,0,stream>>>(s.mu, HDIM);
    k_zero_f<<<1,64,0,stream>>>(s.var, HDIM);
    k_bnstats<<<gb(N*HDIM),256,0,stream>>>(s.h1, s.mu, s.var, N);
    k_bnfin<<<1,64,0,stream>>>(s.mu, s.var, N);
    k_plgate<<<gb(N),256,0,stream>>>(s.h1, s.mu, s.var, P[pp+15], P[pp+16], P[pp+17],
                                     P[pp+18], s.pl, s.lam, s.scal, N, seedGate);
    // seg stats + noisy + kl accumulators
    k_segstat<<<BATCH,128,0,stream>>>(s.f2, s.startB, s.cntB, s.segMean, s.segStd);
    k_noisy<<<gb(N*D2),256,0,stream>>>(s.f2, s.lam, bidx, s.segMean, s.segStd,
                                       s.noisy, s.scal, N, seedNoise);
    // set2set
    k_zero_f<<<gb(BATCH*D2),256,0,stream>>>(s.h_s, BATCH*D2);
    k_zero_f<<<gb(BATCH*D2),256,0,stream>>>(s.c_s, BATCH*D2);
    k_zero_f<<<gb(BATCH*D4),256,0,stream>>>(s.qstar, BATCH*D4);
    for (int it = 0; it < 2; ++it){
      k_s2s_gates<<<gb(BATCH*G4),256,0,stream>>>(s.qstar, s.h_s, P[pp+9], P[pp+11],
                                                 P[pp+10], P[pp+12], gbuf);
      k_s2s_update<<<gb(BATCH*D2),256,0,stream>>>(gbuf, s.h_s, s.c_s);
      k_s2s_e<<<gb(N),256,0,stream>>>(s.noisy, s.h_s, bidx, s.evec, N);
      k_s2s_attn<<<BATCH,128,0,stream>>>(s.noisy, s.evec, s.startB, s.cntB, s.r_s);
      k_s2s_q<<<gb(BATCH*D4),256,0,stream>>>(s.h_s, s.r_s, s.qstar);
    }
  };
  run_bottleneck(bu, Nu, PU, su, 0x2001u, 0x2002u);
  run_bottleneck(bv, Nv, PV, sv, 0x2003u, 0x2004u);

  // ---- head ----
  k_mlp1<<<gb(BATCH*256),256,0,stream>>>(su.qstar, sv.qstar, P[38], P[39], h1m);
  k_dense<<<gb(BATCH*128),256,0,stream>>>(h1m, P[40], P[41], h2m, BATCH, 256, 128, 1);
  k_mlp3_out<<<gb(BATCH),256,0,stream>>>(h2m, P[42], P[43], su.scal, sv.scal,
                                         Nu, Nv, (float*)d_out, out_size);
}
